// MetaExpert_18219251269835
// MI455X (gfx1250) — matrix operand order repaired and checked by reading, not compiled, not run on hardware
//
#include <hip/hip_runtime.h>
#include <hip/hip_fp16.h>
#include <cstdint>

#define S_SITES 4096
#define TOTALN  131072
#define C_IN    12
#define LSEQ    32
#define FDIM    32
#define KW      5
#define NCLS    3
#define KKPAD   64            // C_IN*KW = 60, padded to 64
#define XROW    36            // 32 cols + 2 halo each side (f16)
#define XTILE   480           // 13 rows * 36 + slack, zero padded

typedef __attribute__((ext_vector_type(16))) _Float16 v16h;
typedef __attribute__((ext_vector_type(2)))  _Float16 h2;
typedef __attribute__((ext_vector_type(8)))  float    v8f;
typedef __attribute__((ext_vector_type(4)))  uint32_t u32x4;
typedef __attribute__((ext_vector_type(4)))  int      i32x4;
typedef __attribute__((ext_vector_type(8)))  int      i32x8;

#if defined(__has_builtin)
#  if __has_builtin(__builtin_amdgcn_tensor_load_to_lds) && __has_builtin(__builtin_amdgcn_s_wait_tensorcnt)
#    define HAVE_TDM 1
#  endif
#endif
#ifndef HAVE_TDM
#  define HAVE_TDM 0
#endif

// ---------------------------------------------------------------------------
// TDM: DMA one read's raw f32 tile (384 contiguous elements) into LDS.
// 1-D tensor descriptor: data_size=4B, tile_dim0=384, tensor_dim0=384.
// ---------------------------------------------------------------------------
__device__ __forceinline__ void tdm_issue(const float* gsrc, uint32_t lds_off) {
#if HAVE_TDM
  uint64_t ga = (uint64_t)(uintptr_t)gsrc;
  u32x4 g0;
  g0[0] = 1u;                                               // count=1, flags=0
  g0[1] = lds_off;                                          // LDS byte address
  g0[2] = (uint32_t)ga;                                     // global addr lo
  g0[3] = ((uint32_t)(ga >> 32) & 0x01FFFFFFu) | (2u << 30);// addr[56:32] | type=2
  i32x8 g1;
  g1[0] = (2 << 16);                 // workgroup_mask=0, data_size=4B, no flags
  g1[1] = (384 & 0xFFFF) << 16;      // tensor_dim0[15:0]
  g1[2] = (384 >> 16) | (1 << 16);   // tensor_dim0[31:16] | tensor_dim1[15:0]=1
  g1[3] = (384 << 16);               // tensor_dim1 hi=0 | tile_dim0=384
  g1[4] = 1;                         // tile_dim1=1, tile_dim2=0
  g1[5] = 384;                       // tensor_dim0_stride lo
  g1[6] = (384 << 16);               // stride0 hi=0 | tensor_dim1_stride lo
  g1[7] = 0;
  i32x4 z4 = {0, 0, 0, 0};
#if __clang_major__ >= 23
  i32x8 z8 = {0, 0, 0, 0, 0, 0, 0, 0};
  __builtin_amdgcn_tensor_load_to_lds(g0, g1, z4, z4, z8, 0);
#else
  __builtin_amdgcn_tensor_load_to_lds(g0, g1, z4, z4, 0);
#endif
#else
  (void)gsrc; (void)lds_off;
#endif
}

// A fragment (16x32 f16, contiguous K order, same as gatherB): lane holds M=lane&15;
// lanes 0-15: K = 0..15, lanes 16-31: K = 16..31 (+32*kt)
__device__ __forceinline__ v16h gatherA(const _Float16* Wl, int lane, int mt, int kt) {
  const int f    = mt * 16 + (lane & 15);
  const int aoff = (lane & 16) ? 16 : 0;
  const _Float16* row = Wl + f * KKPAD + kt * 32 + aoff;
  v16h r;
#pragma unroll
  for (int e = 0; e < 16; ++e) {
    int kk = e;
    r[e] = row[kk];
  }
  return r;
}

// B fragment (32x16 f16): lane = column N (mod 16); lanes 0-15 hold K 0..15,
// lanes 16-31 hold K 16..31 (+32*kt). im2col on the fly from padded LDS tile:
// element (K=c*5+k, N=col) = x[c][col + k - 2]; halo/tail rows are zeros.
// Addresses are loop-invariant -> compiler hoists all address math.
__device__ __forceinline__ v16h gatherB(const _Float16* Xl, int lane, int kt, int nt) {
  const int col = nt * 16 + (lane & 15);
  const int kb  = kt * 32 + ((lane & 16) ? 16 : 0);
  v16h r;
#pragma unroll
  for (int e = 0; e < 16; ++e) {
    int kap = kb + e;                // padded K index, 60..63 hit the zero row
    int c   = kap / 5;
    int k   = kap - c * 5;
    r[e] = Xl[c * XROW + col + k];   // == x[c][col + k - 2] with halo offset
  }
  return r;
}

// Cross-lane column reduction (16-lane groups) + atomic segment-sum flush.
// Only called on site transitions (~1 per 32 reads), not per read.
__device__ __forceinline__ void flush_site(float fa[2][8], float* dst, int lane) {
#pragma unroll
  for (int mt = 0; mt < 2; ++mt) {
    float red[8];
#pragma unroll
    for (int r = 0; r < 8; ++r) {
      float v = fa[mt][r];
      v += __shfl_xor(v, 1);
      v += __shfl_xor(v, 2);
      v += __shfl_xor(v, 4);
      v += __shfl_xor(v, 8);
      red[r] = v;
    }
    if ((lane & 15) == 0) {
      int rb = mt * 16 + ((lane & 16) ? 8 : 0);
#pragma unroll
      for (int r = 0; r < 8; ++r)
        __hip_atomic_fetch_add(&dst[rb + r], red[r] * (1.0f / 32.0f),
                               __ATOMIC_RELAXED, __HIP_MEMORY_SCOPE_AGENT);
    }
  }
}

// ---------------------------------------------------------------------------
// Stage 1: per-read conv features via WMMA, segment-summed into acc.
// grid = (512, 2): blockIdx.y selects input; 4096 waves/input, 32 reads/wave.
// ---------------------------------------------------------------------------
__global__ __launch_bounds__(256) void conv_feat_kernel(
    const float* __restrict__ x0, const float* __restrict__ x1,
    const float* __restrict__ W0g, const float* __restrict__ b0g,
    const float* __restrict__ W1g, const float* __restrict__ b1g,
    const int* __restrict__ offs, float* __restrict__ acc) {
  __shared__ _Float16 Wl[FDIM * KKPAD];
  __shared__ float    bl[FDIM];
  __shared__ __align__(16) float stage[8][2][C_IN * LSEQ];
  __shared__ _Float16 xl[8][XTILE];

  const int inp = blockIdx.y;
  const float* xg = inp ? x1 : x0;
  const float* Wg = inp ? W1g : W0g;
  const float* bg = inp ? b1g : b0g;

  // stage weights as f16, K zero-padded 60 -> 64
  for (int idx = threadIdx.x; idx < FDIM * KKPAD; idx += blockDim.x) {
    int f = idx >> 6, kk = idx & 63;
    Wl[idx] = (_Float16)(kk < C_IN * KW ? Wg[f * (C_IN * KW) + kk] : 0.0f);
  }
  if (threadIdx.x < FDIM) bl[threadIdx.x] = bg[threadIdx.x];
  __syncthreads();

  const int lane = threadIdx.x & 31;
  const int wv   = threadIdx.x >> 5;

  // zero the padded f16 tile once: halo columns + phantom row stay zero forever
  for (int idx = lane; idx < XTILE; idx += 32) xl[wv][idx] = (_Float16)0.0f;

  // per-lane bias for the C-tile rows this lane owns (M = r + 8*(lane>=16))
  float bb[2][8];
#pragma unroll
  for (int mt = 0; mt < 2; ++mt)
#pragma unroll
    for (int r = 0; r < 8; ++r)
      bb[mt][r] = bl[mt * 16 + r + ((lane & 16) ? 8 : 0)];

  // weight fragments (reused for every read)
  v16h A[2][2];
#pragma unroll
  for (int mt = 0; mt < 2; ++mt)
#pragma unroll
    for (int kt = 0; kt < 2; ++kt)
      A[mt][kt] = gatherA(Wl, lane, mt, kt);

  const int waves_per_input = gridDim.x * 8;
  const int chunk = TOTALN / waves_per_input;
  const int n0    = (blockIdx.x * 8 + wv) * chunk;
  const int* off  = offs + inp * (S_SITES + 1);

  // one binary search per WAVE (not per read); afterwards advance incrementally
  int site_lo = 0, site_hi = S_SITES;
  while (site_hi - site_lo > 1) {
    int mid = (site_lo + site_hi) >> 1;
    if (off[mid] <= n0) site_lo = mid; else site_hi = mid;
  }
  int site     = site_lo;
  int next_off = off[site + 1];
  float* const acc_base = acc + (size_t)inp * S_SITES * FDIM;

  // per-lane running feature accumulator for the current site
  float fa[2][8];
#pragma unroll
  for (int mt = 0; mt < 2; ++mt)
#pragma unroll
    for (int r = 0; r < 8; ++r) fa[mt][r] = 0.0f;

  tdm_issue(xg + (size_t)n0 * (C_IN * LSEQ), (uint32_t)(uintptr_t)&stage[wv][0][0]);

  for (int i = 0; i < chunk; ++i) {
    const int n = n0 + i;

    // site transition (wave-uniform): flush accumulated features, advance
    if (n >= next_off) {
      flush_site(fa, acc_base + (size_t)site * FDIM, lane);
#pragma unroll
      for (int mt = 0; mt < 2; ++mt)
#pragma unroll
        for (int r = 0; r < 8; ++r) fa[mt][r] = 0.0f;
      do { ++site; } while (off[site + 1] <= n);   // skips empty sites
      next_off = off[site + 1];
    }

#if HAVE_TDM
    if (i + 1 < chunk) {
      tdm_issue(xg + (size_t)(n + 1) * (C_IN * LSEQ),
                (uint32_t)(uintptr_t)&stage[wv][(i + 1) & 1][0]);
      __builtin_amdgcn_s_wait_tensorcnt((short)1);
    } else {
      __builtin_amdgcn_s_wait_tensorcnt((short)0);
    }
    asm volatile("" ::: "memory");
    const float* sp = &stage[wv][i & 1][0];
#pragma unroll
    for (int j = 0; j < 3; ++j) {
      int f4 = lane + j * 32;
      float4 v = reinterpret_cast<const float4*>(sp)[f4];
      int c = f4 >> 3, col = (f4 & 7) * 4;
      _Float16* d = &xl[wv][c * XROW + 2 + col];
      h2 lo = {(_Float16)v.x, (_Float16)v.y};
      h2 hi = {(_Float16)v.z, (_Float16)v.w};
      *reinterpret_cast<h2*>(d)     = lo;
      *reinterpret_cast<h2*>(d + 2) = hi;
    }
#else
    const float* gx = xg + (size_t)n * (C_IN * LSEQ);
#pragma unroll
    for (int j = 0; j < 3; ++j) {
      int f4 = lane + j * 32;
      float4 v = reinterpret_cast<const float4*>(gx)[f4];
      int c = f4 >> 3, col = (f4 & 7) * 4;
      _Float16* d = &xl[wv][c * XROW + 2 + col];
      h2 lo = {(_Float16)v.x, (_Float16)v.y};
      h2 hi = {(_Float16)v.z, (_Float16)v.w};
      *reinterpret_cast<h2*>(d)     = lo;
      *reinterpret_cast<h2*>(d + 2) = hi;
    }
#endif
    asm volatile("s_wait_dscnt 0" ::: "memory");   // f16 tile visible to gathers

    // Y[32x32] = W[32x64] x im2col[64x32] : 8 WMMAs
    v8f c00 = {}, c01 = {}, c10 = {}, c11 = {};
#pragma unroll
    for (int kt = 0; kt < 2; ++kt) {
      v16h B0 = gatherB(&xl[wv][0], lane, kt, 0);
      v16h B1 = gatherB(&xl[wv][0], lane, kt, 1);
      c00 = __builtin_amdgcn_wmma_f32_16x16x32_f16(false, A[0][kt], false, B0,
                                                   (short)0, c00, false, false);
      c01 = __builtin_amdgcn_wmma_f32_16x16x32_f16(false, A[0][kt], false, B1,
                                                   (short)0, c01, false, false);
      c10 = __builtin_amdgcn_wmma_f32_16x16x32_f16(false, A[1][kt], false, B0,
                                                   (short)0, c10, false, false);
      c11 = __builtin_amdgcn_wmma_f32_16x16x32_f16(false, A[1][kt], false, B1,
                                                   (short)0, c11, false, false);
    }

    // bias + ReLU per element, accumulate per-lane (reduction deferred to flush)
#pragma unroll
    for (int mt = 0; mt < 2; ++mt) {
      const v8f& t0 = mt ? c10 : c00;
      const v8f& t1 = mt ? c11 : c01;
#pragma unroll
      for (int r = 0; r < 8; ++r)
        fa[mt][r] += fmaxf(t0[r] + bb[mt][r], 0.0f) +
                     fmaxf(t1[r] + bb[mt][r], 0.0f);
    }
  }

  flush_site(fa, acc_base + (size_t)site * FDIM, lane);
}

// ---------------------------------------------------------------------------
// Exclusive prefix-scan of counts[2][S] -> offs[2][S+1]
// ---------------------------------------------------------------------------
__global__ __launch_bounds__(256) void scan_kernel(const int* __restrict__ counts,
                                                   int* __restrict__ offs) {
  __shared__ int tmp[256];
  const int t = threadIdx.x;
  for (int row = 0; row < 2; ++row) {
    const int* c = counts + row * S_SITES;
    int* o = offs + row * (S_SITES + 1);
    int loc[16];
    int base = t * 16, s = 0;
#pragma unroll
    for (int i = 0; i < 16; ++i) { loc[i] = s; s += c[base + i]; }
    tmp[t] = s;
    __syncthreads();
    for (int d = 1; d < 256; d <<= 1) {
      int v = (t >= d) ? tmp[t - d] : 0;
      __syncthreads();
      tmp[t] += v;
      __syncthreads();
    }
    int pre = t ? tmp[t - 1] : 0;
#pragma unroll
    for (int i = 0; i < 16; ++i) o[base + i] = pre + loc[i];
    if (t == 255) o[S_SITES] = tmp[255];
    __syncthreads();
  }
}

__global__ void zero_kernel(float* p, int n) {
  int i = blockIdx.x * blockDim.x + threadIdx.x;
  if (i < n) p[i] = 0.0f;
}

// ---------------------------------------------------------------------------
// Stage 2: [S, 2F] @ W2^T + b2, softmax
// ---------------------------------------------------------------------------
__global__ __launch_bounds__(128) void head_kernel(const float* __restrict__ acc,
                                                   const float* __restrict__ W2,
                                                   const float* __restrict__ b2,
                                                   float* __restrict__ out) {
  __shared__ float w[NCLS * 2 * FDIM];
  __shared__ float bb[NCLS];
  for (int i = threadIdx.x; i < NCLS * 2 * FDIM; i += blockDim.x) w[i] = W2[i];
  if (threadIdx.x < NCLS) bb[threadIdx.x] = b2[threadIdx.x];
  __syncthreads();
  int s = blockIdx.x * blockDim.x + threadIdx.x;
  if (s >= S_SITES) return;
  float l0 = bb[0], l1 = bb[1], l2 = bb[2];
  const float* r0 = acc + (size_t)s * FDIM;
  const float* r1 = acc + (size_t)(S_SITES + s) * FDIM;
#pragma unroll
  for (int j = 0; j < FDIM; ++j) {
    float v = r0[j];
    l0 += w[0 * 64 + j] * v; l1 += w[1 * 64 + j] * v; l2 += w[2 * 64 + j] * v;
  }
#pragma unroll
  for (int j = 0; j < FDIM; ++j) {
    float v = r1[j];
    l0 += w[0 * 64 + 32 + j] * v; l1 += w[1 * 64 + 32 + j] * v; l2 += w[2 * 64 + 32 + j] * v;
  }
  float m = fmaxf(l0, fmaxf(l1, l2));
  float e0 = __expf(l0 - m), e1 = __expf(l1 - m), e2 = __expf(l2 - m);
  float inv = 1.0f / (e0 + e1 + e2);
  out[s * 3 + 0] = e0 * inv;
  out[s * 3 + 1] = e1 * inv;
  out[s * 3 + 2] = e2 * inv;
}

extern "C" void kernel_launch(void* const* d_in, const int* in_sizes, int n_in,
                              void* d_out, int out_size, void* d_ws, size_t ws_size,
                              hipStream_t stream) {
  (void)in_sizes; (void)n_in; (void)out_size; (void)ws_size;
  const float* x0     = (const float*)d_in[0];
  const float* x1     = (const float*)d_in[1];
  const int*   counts = (const int*)d_in[2];
  const float* W0     = (const float*)d_in[3];
  const float* b0     = (const float*)d_in[4];
  const float* W1     = (const float*)d_in[5];
  const float* b1     = (const float*)d_in[6];
  const float* W2     = (const float*)d_in[7];
  const float* b2     = (const float*)d_in[8];
  float* out = (float*)d_out;

  float* acc  = (float*)d_ws;                                     // 2*S*F f32 = 1 MB
  int*   offs = (int*)((char*)d_ws + (size_t)2 * S_SITES * FDIM * sizeof(float));

  const int nacc = 2 * S_SITES * FDIM;
  zero_kernel<<<(nacc + 255) / 256, 256, 0, stream>>>(acc, nacc);
  scan_kernel<<<1, 256, 0, stream>>>(counts, offs);
  conv_feat_kernel<<<dim3(512, 2), 256, 0, stream>>>(x0, x1, W0, b0, W1, b1, offs, acc);
  head_kernel<<<(S_SITES + 127) / 128, 128, 0, stream>>>(acc, W2, b2, out);
}
